// QLoRABigNet_48163763257523
// MI455X (gfx1250) — matrix operand order repaired and checked by reading, not compiled, not run on hardware
//
#include <hip/hip_runtime.h>
#include <hip/hip_bf16.h>

// ---------------------------------------------------------------------------
// QLoRA BigNet on MI455X (gfx1250, wave32, WMMA + async global->LDS DMA)
//
//  * prep_weights: dequant int4 (groupwise scales) AND fold LoRA (W += lb@la)
//    into one bf16 weight matrix per layer -> 18 x 1024 x 1024 bf16 in ws
//    (37.7 MB, L2-resident across all GEMMs).
//  * activations in bf16 (halves HBM traffic), fp32 accumulate in WMMA.
//  * 18 fused GEMM(+bias+relu / +residual) kernels, v_wmma_f32_16x16x32_bf16,
//    double-buffered LDS tiles fed by GLOBAL_LOAD_ASYNC_TO_LDS_B128
//    (ASYNCcnt-pipelined) when the toolchain exposes the builtin.
//  * layernorm as a wave-per-row pass between blocks.
// ---------------------------------------------------------------------------

typedef __bf16 bf16_t;
typedef __attribute__((ext_vector_type(16))) __bf16 v16bf;
typedef __attribute__((ext_vector_type(8)))  __bf16 v8bf;
typedef __attribute__((ext_vector_type(8)))  float  v8f;

#define C_DIM     1024
#define GEMM_BM   64
#define GEMM_BN   256
#define GEMM_KC   32
#define LDS_STRIDE 40   // 32 + 8 bf16 pad (80B rows: 16B aligned, spreads banks)

#if __has_builtin(__builtin_amdgcn_global_load_async_to_lds_b128) && \
    __has_builtin(__builtin_amdgcn_s_wait_asynccnt)
#define USE_ASYNC_LDS 1
#else
#define USE_ASYNC_LDS 0
#endif

#if USE_ASYNC_LDS
typedef __attribute__((ext_vector_type(4))) int v4i;
typedef __attribute__((address_space(1))) v4i* gv4p;   // global v4i*
typedef __attribute__((address_space(3))) v4i* lv4p;   // LDS v4i*
__device__ __forceinline__ void cp16_async(const bf16_t* g, bf16_t* l)
{
    // per-lane 16B DMA: global -> LDS, tracked by ASYNCcnt
    __builtin_amdgcn_global_load_async_to_lds_b128((gv4p)g, (lv4p)l, 0, 0);
}
#endif

// ---------------------------------------------------------------------------
// Weight prep: W_eff[n][k] = (q - 8) * scale[n][k/16] + sum_r lb[n][r]*la[r][k]
// One thread -> 8 consecutive k of one (layer, n) row. 18*1024*128 threads.
// ---------------------------------------------------------------------------
__global__ void prep_weights(const int* __restrict__ qw,
                             const float* __restrict__ scales,
                             const float* __restrict__ la,
                             const float* __restrict__ lb,
                             bf16_t* __restrict__ Wq)
{
    const long gid = (long)blockIdx.x * blockDim.x + threadIdx.x;
    const int  l   = (int)(gid >> 17);        // / (1024*128)
    const int  n   = (int)((gid >> 7) & 1023);
    const int  k0  = (int)((gid & 127) << 3);

    float accv[8] = {0.f,0.f,0.f,0.f,0.f,0.f,0.f,0.f};
    const float* lbp = lb + ((size_t)l * 1024 + n) * 32;
    const float* lap = la + (size_t)l * 32 * 1024 + k0;
    for (int r = 0; r < 32; ++r) {
        const float bv = lbp[r];
        const float* rowp = lap + (size_t)r * 1024;
        #pragma unroll
        for (int i = 0; i < 8; ++i) accv[i] += bv * rowp[i];
    }

    const size_t base = ((size_t)l * 1024 + n) * 1024 + k0;
    const int*   qp   = qw + base;
    const float  sc   = scales[((size_t)l * 1024 + n) * 64 + (k0 >> 4)];
    bf16_t*      op   = Wq + base;
    #pragma unroll
    for (int i = 0; i < 8; ++i)
        op[i] = (bf16_t)((float)(qp[i] - 8) * sc + accv[i]);
}

// ---------------------------------------------------------------------------
// fp32 -> bf16 input conversion
// ---------------------------------------------------------------------------
__global__ void f32_to_bf16(const float* __restrict__ x, bf16_t* __restrict__ y)
{
    const size_t i = ((size_t)blockIdx.x * blockDim.x + threadIdx.x) * 4;
    #pragma unroll
    for (int j = 0; j < 4; ++j) y[i + j] = (bf16_t)x[i + j];
}

// ---------------------------------------------------------------------------
// WMMA GEMM: out = A @ W^T + bias  (+relu) (+residual)
//   A: [M][1024] bf16 row-major, W: [1024(out)][1024(in)] bf16 row-major.
//   WG = 256 threads (8 waves, 2x4), tile 64(M) x 256(N), K chunk = 32.
// ---------------------------------------------------------------------------
template<bool RELU, bool RES, bool F32OUT>
__global__ __launch_bounds__(256)
void qgemm(const bf16_t* __restrict__ A, const bf16_t* __restrict__ W,
           const float* __restrict__ bias, const bf16_t* __restrict__ resid,
           bf16_t* __restrict__ outb, float* __restrict__ outf)
{
    const int tid  = threadIdx.x;
    const int wgM  = blockIdx.y * GEMM_BM;
    const int wgN  = blockIdx.x * GEMM_BN;

    const int wave   = tid >> 5;
    const int lane   = tid & 31;
    const int hi16   = lane >> 4;         // which half-wave
    const int lm     = lane & 15;
    const int wmBase = (wave >> 2) * 32;  // 0 / 32
    const int wnBase = (wave & 3) * 64;   // 0 / 64 / 128 / 192

    // global->LDS staging: each thread moves one 16B A chunk + four 16B B chunks
    const int ldRow = tid >> 2;           // 0..63
    const int ldK   = (tid & 3) * 8;      // 0,8,16,24

    v8f acc[2][4] = {};

    const bf16_t* gA  = A + (size_t)(wgM + ldRow) * C_DIM + ldK;
    const bf16_t* gB0 = W + (size_t)(wgN + ldRow) * C_DIM + ldK;

#if USE_ASYNC_LDS
    // ---- async-DMA double-buffered pipeline (no VGPR staging) ----
    __shared__ bf16_t sA[2][GEMM_BM * LDS_STRIDE];
    __shared__ bf16_t sB[2][GEMM_BN * LDS_STRIDE];

    // prime buffer 0 with chunk 0 (5 DMAs per thread)
    cp16_async(gA,                &sA[0][ ldRow        * LDS_STRIDE + ldK]);
    cp16_async(gB0,               &sB[0][ ldRow        * LDS_STRIDE + ldK]);
    cp16_async(gB0 +  64 * C_DIM, &sB[0][(ldRow +  64) * LDS_STRIDE + ldK]);
    cp16_async(gB0 + 128 * C_DIM, &sB[0][(ldRow + 128) * LDS_STRIDE + ldK]);
    cp16_async(gB0 + 192 * C_DIM, &sB[0][(ldRow + 192) * LDS_STRIDE + ldK]);

    for (int kc = 0; kc < C_DIM; kc += GEMM_KC) {
        const int cur = (kc >> 5) & 1;
        const int nxt = cur ^ 1;
        if (kc + GEMM_KC < C_DIM) {
            const int kn = kc + GEMM_KC;
            cp16_async(gA + kn,                &sA[nxt][ ldRow        * LDS_STRIDE + ldK]);
            cp16_async(gB0 + kn,               &sB[nxt][ ldRow        * LDS_STRIDE + ldK]);
            cp16_async(gB0 + kn +  64 * C_DIM, &sB[nxt][(ldRow +  64) * LDS_STRIDE + ldK]);
            cp16_async(gB0 + kn + 128 * C_DIM, &sB[nxt][(ldRow + 128) * LDS_STRIDE + ldK]);
            cp16_async(gB0 + kn + 192 * C_DIM, &sB[nxt][(ldRow + 192) * LDS_STRIDE + ldK]);
            // 10 in flight; ASYNC ops complete in order -> <=5 means chunk kc done
            __builtin_amdgcn_s_wait_asynccnt(5);
        } else {
            __builtin_amdgcn_s_wait_asynccnt(0);
        }
        __syncthreads();   // everyone's DMAs for buffer `cur` are visible

        v16bf afrag[2];
        #pragma unroll
        for (int mi = 0; mi < 2; ++mi) {
            const bf16_t* ap = &sA[cur][(wmBase + mi * 16 + lm) * LDS_STRIDE];
            v8bf lo = *(const v8bf*)(ap + hi16 * 8);
            v8bf hh = *(const v8bf*)(ap + 16 + hi16 * 8);
            afrag[mi] = __builtin_shufflevector(lo, hh,
                0,1,2,3,4,5,6,7,8,9,10,11,12,13,14,15);
        }
        #pragma unroll
        for (int ni = 0; ni < 4; ++ni) {
            const bf16_t* bp = &sB[cur][(wnBase + ni * 16 + lm) * LDS_STRIDE];
            v8bf lo = *(const v8bf*)(bp + hi16 * 8);
            v8bf hh = *(const v8bf*)(bp + 16 + hi16 * 8);
            v16bf bfrag = __builtin_shufflevector(lo, hh,
                0,1,2,3,4,5,6,7,8,9,10,11,12,13,14,15);
            #pragma unroll
            for (int mi = 0; mi < 2; ++mi)
                acc[mi][ni] = __builtin_amdgcn_wmma_f32_16x16x32_bf16(
                    false, afrag[mi], false, bfrag,
                    (short)0, acc[mi][ni], false, false);
        }
        __syncthreads();   // reads of buffer `cur` done -> safe to DMA over it
    }
#else
    // ---- fallback: register-staged single-buffer pipeline ----
    __shared__ bf16_t sA1[GEMM_BM * LDS_STRIDE];
    __shared__ bf16_t sB1[GEMM_BN * LDS_STRIDE];

    v8bf ra, rb0, rb1, rb2, rb3;
    ra  = *(const v8bf*)gA;
    rb0 = *(const v8bf*)(gB0);
    rb1 = *(const v8bf*)(gB0 +  64 * C_DIM);
    rb2 = *(const v8bf*)(gB0 + 128 * C_DIM);
    rb3 = *(const v8bf*)(gB0 + 192 * C_DIM);

    for (int kc = 0; kc < C_DIM; kc += GEMM_KC) {
        __syncthreads();
        *(v8bf*)&sA1[ ldRow        * LDS_STRIDE + ldK] = ra;
        *(v8bf*)&sB1[ ldRow        * LDS_STRIDE + ldK] = rb0;
        *(v8bf*)&sB1[(ldRow +  64) * LDS_STRIDE + ldK] = rb1;
        *(v8bf*)&sB1[(ldRow + 128) * LDS_STRIDE + ldK] = rb2;
        *(v8bf*)&sB1[(ldRow + 192) * LDS_STRIDE + ldK] = rb3;
        __syncthreads();

        const int kn = kc + GEMM_KC;
        if (kn < C_DIM) {
            ra  = *(const v8bf*)(gA  + kn);
            rb0 = *(const v8bf*)(gB0 + kn);
            rb1 = *(const v8bf*)(gB0 + kn +  64 * C_DIM);
            rb2 = *(const v8bf*)(gB0 + kn + 128 * C_DIM);
            rb3 = *(const v8bf*)(gB0 + kn + 192 * C_DIM);
        }

        v16bf afrag[2];
        #pragma unroll
        for (int mi = 0; mi < 2; ++mi) {
            const bf16_t* ap = &sA1[(wmBase + mi * 16 + lm) * LDS_STRIDE];
            v8bf lo = *(const v8bf*)(ap + hi16 * 8);
            v8bf hh = *(const v8bf*)(ap + 16 + hi16 * 8);
            afrag[mi] = __builtin_shufflevector(lo, hh,
                0,1,2,3,4,5,6,7,8,9,10,11,12,13,14,15);
        }
        #pragma unroll
        for (int ni = 0; ni < 4; ++ni) {
            const bf16_t* bp = &sB1[(wnBase + ni * 16 + lm) * LDS_STRIDE];
            v8bf lo = *(const v8bf*)(bp + hi16 * 8);
            v8bf hh = *(const v8bf*)(bp + 16 + hi16 * 8);
            v16bf bfrag = __builtin_shufflevector(lo, hh,
                0,1,2,3,4,5,6,7,8,9,10,11,12,13,14,15);
            #pragma unroll
            for (int mi = 0; mi < 2; ++mi)
                acc[mi][ni] = __builtin_amdgcn_wmma_f32_16x16x32_bf16(
                    false, afrag[mi], false, bfrag,
                    (short)0, acc[mi][ni], false, false);
        }
    }
#endif

    // Epilogue: VGPR j of C holds M = j (lanes 0-15) or M = 8+j (lanes 16-31),
    // N = lane % 16.
    #pragma unroll
    for (int mi = 0; mi < 2; ++mi) {
        const int r0 = wgM + wmBase + mi * 16 + hi16 * 8;
        #pragma unroll
        for (int ni = 0; ni < 4; ++ni) {
            const int col = wgN + wnBase + ni * 16 + lm;
            const float bv = bias[col];
            #pragma unroll
            for (int j = 0; j < 8; ++j) {
                float v = acc[mi][ni][j] + bv;
                if (RELU) v = fmaxf(v, 0.0f);
                const size_t idx = (size_t)(r0 + j) * C_DIM + col;
                if (RES) v += (float)resid[idx];
                if (F32OUT) outf[idx] = v;
                else        outb[idx] = (bf16_t)v;
            }
        }
    }
}

// ---------------------------------------------------------------------------
// LayerNorm over last dim (1024): one wave32 per row, 8 rows per WG.
// ---------------------------------------------------------------------------
__global__ void ln_kernel(const bf16_t* __restrict__ in,
                          const float* __restrict__ g, const float* __restrict__ b,
                          bf16_t* __restrict__ out)
{
    const int wave = threadIdx.x >> 5;
    const int lane = threadIdx.x & 31;
    const int row  = blockIdx.x * 8 + wave;

    const bf16_t* p = in + (size_t)row * C_DIM;
    float vals[32];
    float s = 0.f, s2 = 0.f;
    #pragma unroll
    for (int i = 0; i < 32; ++i) {
        const float v = (float)p[lane + i * 32];
        vals[i] = v; s += v; s2 += v * v;
    }
    #pragma unroll
    for (int off = 16; off > 0; off >>= 1) {
        s  += __shfl_xor(s,  off, 32);
        s2 += __shfl_xor(s2, off, 32);
    }
    const float mu  = s  * (1.0f / C_DIM);
    const float var = s2 * (1.0f / C_DIM) - mu * mu;
    const float rs  = rsqrtf(var + 1e-5f);

    bf16_t* o = out + (size_t)row * C_DIM;
    #pragma unroll
    for (int i = 0; i < 32; ++i) {
        const int c = lane + i * 32;
        o[c] = (bf16_t)((vals[i] - mu) * rs * g[c] + b[c]);
    }
}

// ---------------------------------------------------------------------------
// Host launcher
// ---------------------------------------------------------------------------
extern "C" void kernel_launch(void* const* d_in, const int* in_sizes, int n_in,
                              void* d_out, int out_size, void* d_ws, size_t ws_size,
                              hipStream_t stream)
{
    const float* x      = (const float*)d_in[0];
    const int*   qw     = (const int*)  d_in[1];
    const float* scales = (const float*)d_in[2];
    const float* bias   = (const float*)d_in[3];
    const float* lora_a = (const float*)d_in[4];
    const float* lora_b = (const float*)d_in[5];
    const float* ln_g   = (const float*)d_in[6];
    const float* ln_b   = (const float*)d_in[7];
    float* out = (float*)d_out;

    const int    M        = in_sizes[0] / C_DIM;   // 32768
    const size_t actElems = (size_t)M * C_DIM;

    // Workspace layout (bf16): [18 weight matrices][h][tA][tB]
    bf16_t* Wq = (bf16_t*)d_ws;
    bf16_t* hB = Wq + (size_t)18 * C_DIM * C_DIM;
    bf16_t* tA = hB + actElems;
    bf16_t* tB = tA + actElems;

    // 1) dequant + LoRA-fold weights into bf16 (deterministic, reads only d_in)
    prep_weights<<<18 * 1024 * 128 / 256, 256, 0, stream>>>(qw, scales, lora_a, lora_b, Wq);

    // 2) x -> bf16
    f32_to_bf16<<<(unsigned)(actElems / 1024), 256, 0, stream>>>(x, hB);

    // 3) 6 blocks x 3 fused GEMMs (+LN between blocks)
    const dim3 gg(C_DIM / GEMM_BN, M / GEMM_BM);
    for (int blk = 0; blk < 6; ++blk) {
        const bf16_t* w1 = Wq + (size_t)(3 * blk + 0) * C_DIM * C_DIM;
        const bf16_t* w2 = Wq + (size_t)(3 * blk + 1) * C_DIM * C_DIM;
        const bf16_t* w3 = Wq + (size_t)(3 * blk + 2) * C_DIM * C_DIM;
        const float*  b1 = bias + (size_t)(3 * blk + 0) * C_DIM;
        const float*  b2 = bias + (size_t)(3 * blk + 1) * C_DIM;
        const float*  b3 = bias + (size_t)(3 * blk + 2) * C_DIM;

        qgemm<true,  false, false><<<gg, 256, 0, stream>>>(hB, w1, b1, nullptr, tA, nullptr);
        qgemm<true,  false, false><<<gg, 256, 0, stream>>>(tA, w2, b2, nullptr, tB, nullptr);
        if (blk < 5) {
            qgemm<false, true, false><<<gg, 256, 0, stream>>>(tB, w3, b3, hB, tA, nullptr);
            ln_kernel<<<M / 8, 256, 0, stream>>>(tA, ln_g + (size_t)blk * C_DIM,
                                                 ln_b + (size_t)blk * C_DIM, hB);
        } else {
            qgemm<false, true, true><<<gg, 256, 0, stream>>>(tB, w3, b3, hB, nullptr, out);
        }
    }
}
